// TextEncoder_43026982371335
// MI455X (gfx1250) — matrix operand order repaired and checked by reading, not compiled, not run on hardware
//
#include <hip/hip_runtime.h>
#include <hip/hip_bf16.h>

#define DI static __device__ __forceinline__

typedef unsigned short ushort_t;
typedef __attribute__((ext_vector_type(16))) __bf16 v16bf;
typedef __attribute__((ext_vector_type(8)))  float  v8f;
typedef __attribute__((ext_vector_type(4)))  unsigned int v4u;
typedef __attribute__((ext_vector_type(8)))  int v8i;
typedef __attribute__((ext_vector_type(4)))  int v4i;

#define AS1 __attribute__((address_space(1)))
#define AS3 __attribute__((address_space(3)))

// ---- CDNA5 feature detection (compile-safe across toolchains) ----
#if defined(__has_builtin)
# if __has_builtin(__builtin_amdgcn_global_load_async_to_lds_b128)
#  define HAS_ASYNC_LDS 1
# endif
# if __has_builtin(__builtin_amdgcn_s_wait_asynccnt)
#  define HAS_WAIT_ASYNC 1
# endif
# if __has_builtin(__builtin_amdgcn_tensor_load_to_lds) && __has_builtin(__builtin_amdgcn_s_wait_tensorcnt)
#  define HAS_TDM 1
# endif
#endif

// ---- problem constants ----
constexpr int Cc  = 256;
constexpr int Hh  = 4;
constexpr int Ll  = 6;
constexpr int FCc = 1024;
constexpr int OUTc= 192;
constexpr int WSc = 4;
constexpr int HDc = Cc / Hh;           // 64
constexpr int Bb  = 8;
constexpr int Tt  = 1024;
constexpr int MR  = Bb * Tt;           // 8192 rows
constexpr int NREL = 2 * WSc + 1;      // 9

// ---- bf16 helpers ----
DI ushort_t f2bf(float f) {
  unsigned u = __float_as_uint(f);
  u += 0x7FFFu + ((u >> 16) & 1u);     // round to nearest even
  return (ushort_t)(u >> 16);
}
DI float bf2f(ushort_t b) { return __uint_as_float(((unsigned)b) << 16); }

// ---- WMMA fragment helpers (ISA 7.12.2 layouts, wave32) ----
union Frag { v16bf v; uint4 q[2]; };

// A-matrix 16x32 bf16: lane m=l&15; elems 0..7 -> K=half*8+0..7,
// elems 8..15 -> K=16+half*8+0..7
DI v16bf ld_fragA(const ushort_t* p, int half8) {
  Frag f;
  f.q[0] = *(const uint4*)(p + half8);
  f.q[1] = *(const uint4*)(p + 16 + half8);
  return f.v;
}
// B-matrix 32x16 bf16 stored [n][k] in LDS: lane n=l&15; elems 0..7 -> K=half*8+0..7,
// elems 8..15 -> K=16+half*8+0..7
DI v16bf ld_fragB(const ushort_t* p, int half16) {
  Frag f;
  f.q[0] = *(const uint4*)(p + (half16 >> 1));
  f.q[1] = *(const uint4*)(p + 16 + (half16 >> 1));
  return f.v;
}
DI v8f wmma_bf16(v16bf a, v16bf b, v8f c) {
  return __builtin_amdgcn_wmma_f32_16x16x32_bf16(false, a, false, b, (short)0, c,
                                                 false, false);
}
DI v8f zero8() { v8f z = {0.f,0.f,0.f,0.f,0.f,0.f,0.f,0.f}; return z; }

// ---- async global->LDS copy (16B per lane), CDNA5 ASYNCcnt path ----
// prototype (from compiler diagnostic): (v4i AS1* src, v4i AS3* dst, Imm off, Imm cpol)
#if defined(HAS_ASYNC_LDS)
template <int OFS>
DI void async_cp16(const ushort_t* g, ushort_t* l) {
  __builtin_amdgcn_global_load_async_to_lds_b128(
      (AS1 v4i*)g, (AS3 v4i*)l, OFS, 0);
}
DI void async_wait() {
#if defined(HAS_WAIT_ASYNC)
  __builtin_amdgcn_s_wait_asynccnt(0);
#else
  asm volatile("s_wait_asynccnt 0x0" ::: "memory");
#endif
}
#endif

// ---- Tensor Data Mover: 128x32 bf16 tile, LDS row stride 40 ushorts ----
// pad_interval code 3 => pad every 16 DWORDs (64B row), pad_amount code 3 => 4 DWORDs (16B)
#if defined(HAS_TDM)
DI void tdm_tile_128x32(ushort_t* lds, const ushort_t* g, int strideElems) {
  unsigned long long ga = (unsigned long long)(size_t)g;
  unsigned lo = (unsigned)(size_t)(AS3 ushort_t*)lds;   // LDS byte offset
  v4u g0;
  g0[0] = 1u;                                           // count=1, user mode
  g0[1] = lo;                                           // lds_addr
  g0[2] = (unsigned)ga;                                 // global_addr[31:0]
  g0[3] = (unsigned)((ga >> 32) & 0x01FFFFFFu) | (2u << 30);  // [56:32] | type=2
  const unsigned dim0 = 1u << 30, dim1 = 1u << 20;      // huge => no OOB clipping
  v8i g1;
  g1[0] = (int)((1u << 16) | (1u << 20) | (3u << 22) | (3u << 25)); // data_size=2B, pad
  g1[1] = (int)(dim0 << 16);                            // tensor_dim0[15:0] @63:48
  g1[2] = (int)((dim0 >> 16) | (dim1 << 16));           // dim0 hi | dim1 lo
  g1[3] = (int)((dim1 >> 16) | (32u << 16));            // dim1 hi | tile_dim0=32
  g1[4] = 128;                                          // tile_dim1=128, tile_dim2=0
  g1[5] = strideElems;                                  // tensor_dim0_stride[31:0]
  g1[6] = 0;
  g1[7] = 0;
  v4i z4 = {0, 0, 0, 0};
#if __clang_major__ >= 23
  v8i z8 = {0, 0, 0, 0, 0, 0, 0, 0};
  __builtin_amdgcn_tensor_load_to_lds(g0, g1, z4, z4, z8, 0);
#else
  __builtin_amdgcn_tensor_load_to_lds(g0, g1, z4, z4, 0);
#endif
}
#endif

// =====================================================================
// casts: plain, and batched transpose (W[K][N] -> Wt[N][K]) per layer
// =====================================================================
__global__ void k_cast(const float* __restrict__ s, ushort_t* __restrict__ d, int n) {
  int i = blockIdx.x * 256 + threadIdx.x;
  if (i < n) d[i] = f2bf(s[i]);
}
__global__ void k_castT(const float* __restrict__ s, ushort_t* __restrict__ d,
                        int K, int N) {
  int i = blockIdx.x * 256 + threadIdx.x;     // over K*N
  size_t base = (size_t)blockIdx.y * K * N;   // layer
  int k = i / N, n = i - k * N;
  d[base + (size_t)n * K + k] = f2bf(s[base + i]);
}

// =====================================================================
// embedding gather + sqrt(C) scale
// =====================================================================
__global__ void k_embed(const float* __restrict__ emb, const int* __restrict__ toks,
                        float* __restrict__ x, ushort_t* __restrict__ xbf) {
  int idx = blockIdx.x * 256 + threadIdx.x;   // over MR*C
  int row = idx >> 8, c = idx & 255;
  float v = emb[(size_t)toks[row] * Cc + c] * 16.0f;   // sqrt(256)=16
  x[idx] = v;
  xbf[idx] = f2bf(v);
}

// =====================================================================
// bf16 WMMA GEMM:  Y[M,N] = act( (A[M,K] @ Wt[N,K]^T) + bias ) * scale
// block tile 128x128, 8 waves (4x2), wave = 32x64 (2x4 tiles), BK=32.
// Double-buffered LDS: TDM / async-LDS loads for tile t+1 overlap WMMA on t.
// vtrans: write bf16 result transposed per attention head: [(b*H+h)*HD+d][T]
// =====================================================================
__global__ __launch_bounds__(256) void k_gemm(
    const ushort_t* __restrict__ A, const ushort_t* __restrict__ Bw,
    const float* __restrict__ bias, float scale, int do_relu, int vtrans,
    ushort_t* __restrict__ outb, float* __restrict__ outf, int N, int K)
{
  __shared__ __align__(16) ushort_t Alds[2][128][40];   // [buf][m][k] (+16B pad)
  __shared__ __align__(16) ushort_t Blds[2][128][40];   // [buf][n][k] (+16B pad)

  const int tid   = threadIdx.x;
  const int lane  = tid & 31;
  const int wave  = tid >> 5;
  const int half8 = (lane >> 4) * 8;
  const int half16= (lane >> 4) * 16;
  const int ln    = lane & 15;
  const int bm0   = blockIdx.y * 128;
  const int bn0   = blockIdx.x * 128;
  const int wm0   = (wave >> 1) * 32;
  const int wn0   = (wave & 1) * 64;

  auto load_tiles = [&](int buf, int k0) {
#if defined(HAS_TDM)
    if (wave == 0) {
      tdm_tile_128x32(&Alds[buf][0][0], A  + (size_t)bm0 * K + k0, K);
      tdm_tile_128x32(&Blds[buf][0][0], Bw + (size_t)bn0 * K + k0, K);
    }
#else
    int r = tid >> 1, c = (tid & 1) * 16;
    const ushort_t* asrc = A  + (size_t)(bm0 + r) * K + k0 + c;
    const ushort_t* bsrc = Bw + (size_t)(bn0 + r) * K + k0 + c;
#if defined(HAS_ASYNC_LDS)
    async_cp16<0>(asrc, &Alds[buf][r][c]);  async_cp16<16>(asrc, &Alds[buf][r][c]);
    async_cp16<0>(bsrc, &Blds[buf][r][c]);  async_cp16<16>(bsrc, &Blds[buf][r][c]);
#else
    *(uint4*)&Alds[buf][r][c]     = *(const uint4*)(asrc);
    *(uint4*)&Alds[buf][r][c + 8] = *(const uint4*)(asrc + 8);
    *(uint4*)&Blds[buf][r][c]     = *(const uint4*)(bsrc);
    *(uint4*)&Blds[buf][r][c + 8] = *(const uint4*)(bsrc + 8);
#endif
#endif
  };
  auto wait_tiles = [&]() {
#if defined(HAS_TDM)
    if (wave == 0) __builtin_amdgcn_s_wait_tensorcnt(0);
#elif defined(HAS_ASYNC_LDS)
    async_wait();
#endif
  };

  v8f acc[2][4];
  #pragma unroll
  for (int i = 0; i < 2; i++)
    #pragma unroll
    for (int j = 0; j < 4; j++) acc[i][j] = zero8();

  load_tiles(0, 0);
  wait_tiles();
  __syncthreads();

  int cur = 0;
  for (int k0 = 0; k0 < K; k0 += 32) {
    if (k0 + 32 < K) load_tiles(cur ^ 1, k0 + 32);   // overlap DMA with compute

    v16bf af[2], bf_[4];
    #pragma unroll
    for (int mi = 0; mi < 2; mi++) af[mi]  = ld_fragA(&Alds[cur][wm0 + mi * 16 + ln][0], half8);
    #pragma unroll
    for (int ni = 0; ni < 4; ni++) bf_[ni] = ld_fragB(&Blds[cur][wn0 + ni * 16 + ln][0], half16);
    #pragma unroll
    for (int mi = 0; mi < 2; mi++)
      #pragma unroll
      for (int ni = 0; ni < 4; ni++)
        acc[mi][ni] = wmma_bf16(af[mi], bf_[ni], acc[mi][ni]);

    wait_tiles();
    __syncthreads();
    cur ^= 1;
  }

  // bias / scale / relu in place (C layout: elem r -> row r+half8, col ln)
  #pragma unroll
  for (int mi = 0; mi < 2; mi++)
    #pragma unroll
    for (int ni = 0; ni < 4; ni++) {
      float bsv = bias[bn0 + wn0 + ni * 16 + ln];
      #pragma unroll
      for (int r = 0; r < 8; r++) {
        float v = (acc[mi][ni][r] + bsv) * scale;
        if (do_relu) v = fmaxf(v, 0.0f);
        acc[mi][ni][r] = v;
      }
    }

  // uniform store phases
  if (outf != nullptr) {
    #pragma unroll
    for (int mi = 0; mi < 2; mi++)
      #pragma unroll
      for (int ni = 0; ni < 4; ni++) {
        int col = bn0 + wn0 + ni * 16 + ln;
        #pragma unroll
        for (int r = 0; r < 8; r++) {
          int row = bm0 + wm0 + mi * 16 + r + half8;
          outf[(size_t)row * N + col] = acc[mi][ni][r];
        }
      }
  }
  if (outb != nullptr) {
    if (vtrans) {
      #pragma unroll
      for (int mi = 0; mi < 2; mi++)
        #pragma unroll
        for (int ni = 0; ni < 4; ni++) {
          int col = bn0 + wn0 + ni * 16 + ln;     // h*64 + d
          #pragma unroll
          for (int r = 0; r < 8; r++) {
            int row = bm0 + wm0 + mi * 16 + r + half8;   // b*T + t
            size_t dst = ((size_t)((row >> 10) * Hh + (col >> 6)) * HDc + (col & 63)) * Tt
                       + (row & (Tt - 1));
            outb[dst] = f2bf(acc[mi][ni][r]);
          }
        }
    } else {
      #pragma unroll
      for (int mi = 0; mi < 2; mi++)
        #pragma unroll
        for (int ni = 0; ni < 4; ni++) {
          int col = bn0 + wn0 + ni * 16 + ln;
          #pragma unroll
          for (int r = 0; r < 8; r++) {
            int row = bm0 + wm0 + mi * 16 + r + half8;
            outb[(size_t)row * N + col] = f2bf(acc[mi][ni][r]);
          }
        }
    }
  }
}

// =====================================================================
// flash attention, banded relative position (window WS=4)
// grid: (B*H, T/128); block 256 (8 waves); wave owns 16 q rows.
// v pre-transposed: vT[(b*H+h)*HD + d][T].  K/V tiles double-buffered with
// async-LDS; the P round-trip buffer aliases q_lds (q dead after Rq+frags).
// =====================================================================
__global__ __launch_bounds__(256) void k_attn(
    const ushort_t* __restrict__ qg, const ushort_t* __restrict__ kg,
    const ushort_t* __restrict__ vT, const float* __restrict__ relk,
    const float* __restrict__ relv, const int* __restrict__ lens,
    ushort_t* __restrict__ outb)
{
  __shared__ __align__(16) ushort_t q_lds[128][72];     // q tile; later: per-wave P
  __shared__ __align__(16) ushort_t k_lds[2][64][72];   // [buf][key][hd]
  __shared__ __align__(16) ushort_t vT_lds[2][64][72];  // [buf][hd][key]
  __shared__ float Rq[128][NREL];
  __shared__ float rv_lds[NREL][64];

  const int tid    = threadIdx.x;
  const int lane   = tid & 31;
  const int wave   = tid >> 5;
  const int half8  = (lane >> 4) * 8;
  const int half16 = (lane >> 4) * 16;
  const int ln     = lane & 15;
  const int bh = blockIdx.x, b = bh >> 2, h = bh & 3;
  const int q0 = blockIdx.y * 128;
  const int len = lens[b];
  const size_t baseqk = ((size_t)b * Tt) * Cc + h * HDc;
  const size_t basevT = (size_t)(b * Hh + h) * HDc * Tt;
  const int wrow0 = wave * 16;

  auto load_kv = [&](int buf, int kt0) {
    int rr = tid >> 2, c0 = (tid & 3) * 16;   // rr: key for K tile, hd for V tile
    const ushort_t* ks = kg + baseqk + (size_t)(kt0 + rr) * Cc + c0;
    const ushort_t* vs = vT + basevT + (size_t)rr * Tt + kt0 + c0;
#if defined(HAS_ASYNC_LDS)
    async_cp16<0>(ks, &k_lds[buf][rr][c0]);   async_cp16<16>(ks, &k_lds[buf][rr][c0]);
    async_cp16<0>(vs, &vT_lds[buf][rr][c0]);  async_cp16<16>(vs, &vT_lds[buf][rr][c0]);
    if (kt0 + 64 < Tt) { __builtin_prefetch(ks + Cc * 64, 0, 1); __builtin_prefetch(vs + 64, 0, 1); }
#else
    *(uint4*)&k_lds[buf][rr][c0]      = *(const uint4*)(ks);
    *(uint4*)&k_lds[buf][rr][c0 + 8]  = *(const uint4*)(ks + 8);
    *(uint4*)&vT_lds[buf][rr][c0]     = *(const uint4*)(vs);
    *(uint4*)&vT_lds[buf][rr][c0 + 8] = *(const uint4*)(vs + 8);
#endif
  };

  // phase 0: q tile + k/v tile 0 + rel_v
  {
    int r = tid >> 1, c0 = (tid & 1) * 32;
    const ushort_t* src = qg + baseqk + (size_t)(q0 + r) * Cc + c0;
    ushort_t* dst = &q_lds[r][c0];
#if defined(HAS_ASYNC_LDS)
    async_cp16<0>(src, dst);  async_cp16<16>(src, dst);
    async_cp16<32>(src, dst); async_cp16<48>(src, dst);
#else
    #pragma unroll
    for (int j = 0; j < 4; j++)
      *(uint4*)(dst + j * 8) = *(const uint4*)(src + j * 8);
#endif
  }
  load_kv(0, 0);
  for (int idx = tid; idx < NREL * 64; idx += 256)
    rv_lds[idx >> 6][idx & 63] = relv[idx];
#if defined(HAS_ASYNC_LDS)
  async_wait();
#endif
  __syncthreads();

  // Rq[row][j] = q[row,:] . rel_k[j,:]   (q pre-scaled by HD^-0.5)
  for (int idx = tid; idx < 128 * NREL; idx += 256) {
    int row = idx / NREL, j = idx - row * NREL;
    float s = 0.f;
    #pragma unroll
    for (int d = 0; d < HDc; d++) s += bf2f(q_lds[row][d]) * relk[j * HDc + d];
    Rq[row][j] = s;
  }
  v16bf qa0 = ld_fragA(&q_lds[wrow0 + ln][0],  half8);
  v16bf qa1 = ld_fragA(&q_lds[wrow0 + ln][32], half8);
  __syncthreads();   // after this, q_lds is reused as the P buffer

  v8f o_acc[4];
  #pragma unroll
  for (int ni = 0; ni < 4; ni++) o_acc[ni] = zero8();
  float m_run[8], l_run[8];
  #pragma unroll
  for (int r = 0; r < 8; r++) { m_run[r] = -1e30f; l_run[r] = 0.f; }

  int cur = 0;
  for (int kt = 0; kt < Tt / 64; kt++) {
    const int kt0 = kt * 64;
    if (kt0 + 64 < Tt) load_kv(cur ^ 1, kt0 + 64);   // overlap with compute

    // S = q (16x64) @ k^T
    v8f s[4];
    #pragma unroll
    for (int ni = 0; ni < 4; ni++) {
      v16bf kb0 = ld_fragB(&k_lds[cur][ni * 16 + ln][0],  half16);
      v16bf kb1 = ld_fragB(&k_lds[cur][ni * 16 + ln][32], half16);
      s[ni] = wmma_bf16(qa1, kb1, wmma_bf16(qa0, kb0, zero8()));
    }

    const bool band = (kt0 + 63 >= q0 - WSc) && (kt0 <= q0 + 127 + WSc);

    #pragma unroll
    for (int ni = 0; ni < 4; ni++) {
      int colg = kt0 + ni * 16 + ln;
      bool oob = (colg >= len);
      #pragma unroll
      for (int r = 0; r < 8; r++) {
        int rowl = wrow0 + r + half8;
        if (band) {
          int dlt = colg - (q0 + rowl);
          if (dlt >= -WSc && dlt <= WSc) s[ni][r] += Rq[rowl][dlt + WSc];
        }
        if (oob) s[ni][r] = -10000.0f;
      }
    }

    // online softmax
    #pragma unroll
    for (int r = 0; r < 8; r++) {
      float mx = fmaxf(fmaxf(s[0][r], s[1][r]), fmaxf(s[2][r], s[3][r]));
      #pragma unroll
      for (int off = 1; off < 16; off <<= 1) mx = fmaxf(mx, __shfl_xor(mx, off, 32));
      float mn = fmaxf(m_run[r], mx);
      float scl = __expf(m_run[r] - mn);
      float rs = 0.f;
      #pragma unroll
      for (int ni = 0; ni < 4; ni++) {
        float p = __expf(s[ni][r] - mn);
        s[ni][r] = p;
        rs += p;
      }
      #pragma unroll
      for (int off = 1; off < 16; off <<= 1) rs += __shfl_xor(rs, off, 32);
      l_run[r] = l_run[r] * scl + rs;
      m_run[r] = mn;
      #pragma unroll
      for (int ni = 0; ni < 4; ni++) o_acc[ni][r] *= scl;
    }

    // P -> per-wave rows of q_lds (bf16), reload as A fragments
    #pragma unroll
    for (int ni = 0; ni < 4; ni++)
      #pragma unroll
      for (int r = 0; r < 8; r++)
        q_lds[wrow0 + r + half8][ni * 16 + ln] = f2bf(s[ni][r]);

    v16bf pa0 = ld_fragA(&q_lds[wrow0 + ln][0],  half8);
    v16bf pa1 = ld_fragA(&q_lds[wrow0 + ln][32], half8);
    #pragma unroll
    for (int ni = 0; ni < 4; ni++) {
      v16bf vb0 = ld_fragB(&vT_lds[cur][ni * 16 + ln][0],  half16);
      v16bf vb1 = ld_fragB(&vT_lds[cur][ni * 16 + ln][32], half16);
      o_acc[ni] = wmma_bf16(pa1, vb1, wmma_bf16(pa0, vb0, o_acc[ni]));
    }

    // banded rel_v
    if (band) {
      #pragma unroll
      for (int r = 0; r < 8; r++) {
        int rowl = wrow0 + r + half8;
        int rowg = q0 + rowl;
        float pb[NREL];
        #pragma unroll
        for (int j = 0; j < NREL; j++) {
          int cl = rowg + j - WSc - kt0;
          pb[j] = (cl >= 0 && cl < 64) ? bf2f(q_lds[rowl][cl]) : 0.f;
        }
        #pragma unroll
        for (int ni = 0; ni < 4; ni++) {
          int d = ni * 16 + ln;
          float a = 0.f;
          #pragma unroll
          for (int j = 0; j < NREL; j++) a += pb[j] * rv_lds[j][d];
          o_acc[ni][r] += a;
        }
      }
    }

#if defined(HAS_ASYNC_LDS)
    async_wait();
#endif
    __syncthreads();
    cur ^= 1;
  }

  #pragma unroll
  for (int r = 0; r < 8; r++) {
    float inv = 1.0f / l_run[r];
    int rowl = wrow0 + r + half8;
    size_t base = baseqk + (size_t)(q0 + rowl) * Cc;
    #pragma unroll
    for (int ni = 0; ni < 4; ni++)
      outb[base + ni * 16 + ln] = f2bf(o_acc[ni][r] * inv);
  }
}

// =====================================================================
// residual add + layernorm (+optional length mask)
// =====================================================================
__global__ __launch_bounds__(256) void k_res_ln(
    float* __restrict__ x, const float* __restrict__ y,
    const float* __restrict__ g, const float* __restrict__ bta,
    ushort_t* __restrict__ xbf, const int* __restrict__ lens, int apply_mask)
{
  __shared__ float s1[256], s2[256];
  int row = blockIdx.x, c = threadIdx.x;
  size_t idx = (size_t)row * Cc + c;
  float v = x[idx] + y[idx];
  s1[c] = v; s2[c] = v * v;
  __syncthreads();
  for (int off = 128; off > 0; off >>= 1) {
    if (c < off) { s1[c] += s1[c + off]; s2[c] += s2[c + off]; }
    __syncthreads();
  }
  float mean = s1[0] * (1.0f / Cc);
  float var  = s2[0] * (1.0f / Cc) - mean * mean;
  float r = (v - mean) * rsqrtf(var + 1e-5f) * g[c] + bta[c];
  if (apply_mask) {
    int t = row & (Tt - 1), bi = row >> 10;
    if (t >= lens[bi]) r = 0.0f;
  }
  x[idx] = r;
  xbf[idx] = f2bf(r);
}

// =====================================================================
// output scatter kernels
// =====================================================================
__global__ void k_out_xbct(const float* __restrict__ x, float* __restrict__ out) {
  int i = blockIdx.x * 256 + threadIdx.x;        // over B*C*T
  int b = i / (Cc * Tt); int rem = i - b * Cc * Tt;
  int c = rem / Tt; int t = rem - c * Tt;
  out[i] = x[((size_t)b * Tt + t) * Cc + c];
}
__global__ void k_out_stats(const float* __restrict__ y, const int* __restrict__ lens,
                            float* __restrict__ out_m, float* __restrict__ out_s) {
  int i = blockIdx.x * 256 + threadIdx.x;        // over B*384*T
  int b = i / (2 * OUTc * Tt); int rem = i - b * 2 * OUTc * Tt;
  int o = rem / Tt; int t = rem - o * Tt;
  float v = y[((size_t)b * Tt + t) * (2 * OUTc) + o];
  if (t >= lens[b]) v = 0.0f;
  if (o < OUTc) out_m[((size_t)b * OUTc + o) * Tt + t] = v;
  else          out_s[((size_t)b * OUTc + (o - OUTc)) * Tt + t] = v;
}
__global__ void k_out_mask(const int* __restrict__ lens, float* __restrict__ out) {
  int i = blockIdx.x * 256 + threadIdx.x;        // over B*T
  int b = i >> 10, t = i & (Tt - 1);
  out[i] = (t < lens[b]) ? 1.0f : 0.0f;
}

// =====================================================================
// host launcher
// =====================================================================
extern "C" void kernel_launch(void* const* d_in, const int* in_sizes, int n_in,
                              void* d_out, int out_size, void* d_ws, size_t ws_size,
                              hipStream_t stream)
{
  (void)in_sizes; (void)n_in; (void)out_size; (void)ws_size;

  const float* emb    = (const float*)d_in[0];
  const float* Wq     = (const float*)d_in[1];
  const float* Wk     = (const float*)d_in[2];
  const float* Wv     = (const float*)d_in[3];
  const float* Wo     = (const float*)d_in[4];
  const float* bq     = (const float*)d_in[5];
  const float* bk     = (const float*)d_in[6];
  const float* bv     = (const float*)d_in[7];
  const float* bo     = (const float*)d_in[8];
  const float* rel_k  = (const float*)d_in[9];
  const float* rel_v  = (const float*)d_in[10];
  const float* ln1_g  = (const float*)d_in[11];
  const float* ln1_b  = (const float*)d_in[12];
  const float* ln2_g  = (const float*)d_in[13];
  const float* ln2_b  = (const float*)d_in[14];
  const float* ffn_w1 = (const float*)d_in[15];
  const float* ffn_b1 = (const float*)d_in[16];
  const float* ffn_w2 = (const float*)d_in[17];
  const float* ffn_b2 = (const float*)d_in[18];
  const float* proj_w = (const float*)d_in[19];
  const float* proj_b = (const float*)d_in[20];
  const int*   tokens = (const int*)d_in[21];
  const int*   lens   = (const int*)d_in[22];

  char* ws = (char*)d_ws;
  size_t off = 0;
  auto take = [&](size_t bytes) { char* p = ws + off; off += (bytes + 255) & ~size_t(255); return p; };
  float*    x_f    = (float*)   take((size_t)MR * Cc * 4);
  ushort_t* x_bf   = (ushort_t*)take((size_t)MR * Cc * 2);
  ushort_t* q_bf   = (ushort_t*)take((size_t)MR * Cc * 2);
  ushort_t* k_bf   = (ushort_t*)take((size_t)MR * Cc * 2);
  ushort_t* vT_bf  = (ushort_t*)take((size_t)MR * Cc * 2);   // [(b*H+h)*HD+d][T]
  ushort_t* a_bf   = (ushort_t*)take((size_t)MR * Cc * 2);
  float*    y_tmp  = (float*)   take((size_t)MR * Cc * 4);
  ushort_t* h_bf   = (ushort_t*)take((size_t)MR * FCc * 2);
  float*    y_proj = (float*)   take((size_t)MR * 2 * OUTc * 4);
  ushort_t* wq_bf  = (ushort_t*)take((size_t)Ll * Cc * Cc * 2);
  ushort_t* wk_bf  = (ushort_t*)take((size_t)Ll * Cc * Cc * 2);
  ushort_t* wv_bf  = (ushort_t*)take((size_t)Ll * Cc * Cc * 2);
  ushort_t* wo_bf  = (ushort_t*)take((size_t)Ll * Cc * Cc * 2);
  ushort_t* w1_bf  = (ushort_t*)take((size_t)Ll * Cc * FCc * 2);
  ushort_t* w2_bf  = (ushort_t*)take((size_t)Ll * FCc * Cc * 2);
  ushort_t* pT_bf  = (ushort_t*)take((size_t)Cc * 2 * OUTc * 2);

  // weights -> bf16, transposed to [N][K]
  k_castT<<<dim3((Cc * Cc) / 256, Ll), 256, 0, stream>>>(Wq, wq_bf, Cc, Cc);
  k_castT<<<dim3((Cc * Cc) / 256, Ll), 256, 0, stream>>>(Wk, wk_bf, Cc, Cc);
  k_castT<<<dim3((Cc * Cc) / 256, Ll), 256, 0, stream>>>(Wv, wv_bf, Cc, Cc);
  k_castT<<<dim3((Cc * Cc) / 256, Ll), 256, 0, stream>>>(Wo, wo_bf, Cc, Cc);
  k_castT<<<dim3((Cc * FCc) / 256, Ll), 256, 0, stream>>>(ffn_w1, w1_bf, Cc, FCc);
  k_castT<<<dim3((FCc * Cc) / 256, Ll), 256, 0, stream>>>(ffn_w2, w2_bf, FCc, Cc);
  k_cast<<<(2 * OUTc * Cc) / 256, 256, 0, stream>>>(proj_w, pT_bf, 2 * OUTc * Cc);  // already [N][K]

  k_embed<<<(MR * Cc) / 256, 256, 0, stream>>>(emb, tokens, x_f, x_bf);

  const dim3 blk(256);
  const dim3 gC(Cc / 128, MR / 128);
  const dim3 gF(FCc / 128, MR / 128);
  const dim3 gP((2 * OUTc) / 128, MR / 128);

  for (int i = 0; i < Ll; i++) {
    const ushort_t* wq_i = wq_bf + (size_t)i * Cc * Cc;
    const ushort_t* wk_i = wk_bf + (size_t)i * Cc * Cc;
    const ushort_t* wv_i = wv_bf + (size_t)i * Cc * Cc;
    const ushort_t* wo_i = wo_bf + (size_t)i * Cc * Cc;
    const ushort_t* w1_i = w1_bf + (size_t)i * Cc * FCc;
    const ushort_t* w2_i = w2_bf + (size_t)i * FCc * Cc;

    k_gemm<<<gC, blk, 0, stream>>>(x_bf, wq_i, bq + i * Cc, 0.125f, 0, 0, q_bf, nullptr, Cc, Cc);
    k_gemm<<<gC, blk, 0, stream>>>(x_bf, wk_i, bk + i * Cc, 1.0f,   0, 0, k_bf, nullptr, Cc, Cc);
    k_gemm<<<gC, blk, 0, stream>>>(x_bf, wv_i, bv + i * Cc, 1.0f,   0, 1, vT_bf, nullptr, Cc, Cc);

    k_attn<<<dim3(Bb * Hh, Tt / 128), blk, 0, stream>>>(
        q_bf, k_bf, vT_bf, rel_k + (size_t)i * NREL * HDc,
        rel_v + (size_t)i * NREL * HDc, lens, a_bf);

    k_gemm<<<gC, blk, 0, stream>>>(a_bf, wo_i, bo + i * Cc, 1.0f, 0, 0, nullptr, y_tmp, Cc, Cc);
    k_res_ln<<<MR, blk, 0, stream>>>(x_f, y_tmp, ln1_g + i * Cc, ln1_b + i * Cc, x_bf, lens, 0);

    k_gemm<<<gF, blk, 0, stream>>>(x_bf, w1_i, ffn_b1 + i * FCc, 1.0f, 1, 0, h_bf, nullptr, FCc, Cc);
    k_gemm<<<gC, blk, 0, stream>>>(h_bf, w2_i, ffn_b2 + i * Cc, 1.0f, 0, 0, nullptr, y_tmp, Cc, FCc);
    k_res_ln<<<MR, blk, 0, stream>>>(x_f, y_tmp, ln2_g + i * Cc, ln2_b + i * Cc, x_bf, lens, 1);
  }

  k_gemm<<<gP, blk, 0, stream>>>(x_bf, pT_bf, proj_b, 1.0f, 0, 0, nullptr, y_proj, 2 * OUTc, Cc);

  float* out  = (float*)d_out;
  float* out_m = out + (size_t)Bb * Cc * Tt;
  float* out_s = out_m + (size_t)Bb * OUTc * Tt;
  float* out_k = out_s + (size_t)Bb * OUTc * Tt;
  k_out_xbct <<<(Bb * Cc * Tt) / 256, 256, 0, stream>>>(x_f, out);
  k_out_stats<<<(Bb * 2 * OUTc * Tt) / 256, 256, 0, stream>>>(y_proj, lens, out_m, out_s);
  k_out_mask <<<(Bb * Tt) / 256, 256, 0, stream>>>(lens, out_k);
}
